// Decoder_47863115546803
// MI455X (gfx1250) — matrix operand order repaired and checked by reading, not compiled, not run on hardware
//
#include <hip/hip_runtime.h>

// ---------------------------------------------------------------- types ----
typedef __bf16 bf16;
typedef __attribute__((ext_vector_type(16))) __bf16 v16bf;
typedef __attribute__((ext_vector_type(8)))  __bf16 v8bf;
typedef __attribute__((ext_vector_type(8)))  float  v8f;

union V16 { v16bf v; v8bf h[2]; };

__device__ __forceinline__ v8f wmma_bf16(v16bf a, v16bf b, v8f c) {
  return __builtin_amdgcn_wmma_f32_16x16x32_bf16(false, a, false, b, (short)0, c,
                                                 false, false);
}

// Async global->LDS 16B copy (gfx1250 GLOBAL_LOAD_ASYNC_TO_LDS_B128, GV mode).
__device__ __forceinline__ void async_ld16(unsigned lds_off, unsigned long long ga) {
  asm volatile("global_load_async_to_lds_b128 %0, %1, off"
               :: "v"(lds_off), "v"(ga) : "memory");
}
#define WAIT_ASYNC(n) asm volatile("s_wait_asynccnt %0" :: "n"(n) : "memory")

// DPP16 lane-reductions over 16-lane halves (VALU only; keeps DS pipe free
// for WMMA operand ds_loads). 0xB1=quad_perm xor1, 0x4E=quad_perm xor2,
// 0x141=row_half_mirror, 0x140=row_mirror.
template <int C>
__device__ __forceinline__ float dppmv(float x) {
  return __builtin_bit_cast(
      float, __builtin_amdgcn_update_dpp(0, __builtin_bit_cast(int, x), C, 0xf,
                                         0xf, true));
}
__device__ __forceinline__ float redmax16(float x) {
  x = fmaxf(x, dppmv<0xB1>(x));
  x = fmaxf(x, dppmv<0x4E>(x));
  x = fmaxf(x, dppmv<0x141>(x));
  x = fmaxf(x, dppmv<0x140>(x));
  return x;
}
__device__ __forceinline__ float redsum16(float x) {
  x += dppmv<0xB1>(x);
  x += dppmv<0x4E>(x);
  x += dppmv<0x141>(x);
  x += dppmv<0x140>(x);
  return x;
}

#define D_MODEL 512
#define NHEAD   8
#define DH      64
#define BB      4
#define LQ      1024
#define LK      2048
#define LQB     (LQ / 8)   // packed-mask bytes per key

// ------------------------------------------------------------ GEMM --------
// C[M x N] = A[M x K] * Bt[N x K]^T.  Bt tiles (64 n x 64 k) are staged in
// LDS with double-buffered async global->LDS copies shared by all 8 waves.
// mode: 0 = row-major out; 1 = head-split bf16 out [z][h][m][dh];
//       2 = head-split transposed bf16 out [z][h][dh][m].
template <bool AF32>
__global__ __launch_bounds__(256) void k_gemm(
    const void* __restrict__ Aall, const bf16* __restrict__ BtAll,
    const float* __restrict__ bias, float* __restrict__ outF,
    bf16* __restrict__ outB, int M, int N, int K, long sA, long sBt, long sO,
    int mode, int relu, float alpha) {
  __shared__ bf16 Bs[2][64][64];
  const int lane = threadIdx.x & 31;
  const int w    = threadIdx.x >> 5;
  const int r    = lane & 15;
  const int half = lane >> 4;
  const int m0   = blockIdx.y * 128 + w * 16;
  const int n0   = blockIdx.x * 64;
  const int z    = blockIdx.z;
  const bf16* Bt = BtAll + (size_t)z * sBt;

  auto issue_b = [&](int k0, int buf) {
    const int t   = threadIdx.x;
    const int row = t >> 2;
    const int c   = (t & 3) * 32;  // byte offset within 128B row
    const unsigned base = (unsigned)(uintptr_t)(&Bs[buf][0][0]);
    const unsigned long long g =
        (unsigned long long)(uintptr_t)(Bt + (size_t)(n0 + row) * K + k0);
    async_ld16(base + row * 128 + c,      g + c);
    async_ld16(base + row * 128 + c + 16, g + c + 16);
  };

  v8f acc[4] = {};
  const size_t arow = (size_t)(m0 + r) * K;

  issue_b(0, 0);
  for (int k0 = 0; k0 < K; k0 += 64) {
    const int cur = (k0 >> 6) & 1;
    if (k0 + 64 < K) {
      issue_b(k0 + 64, cur ^ 1);
      WAIT_ASYNC(2);   // current tile landed; next tile's 2 ops in flight
    } else {
      WAIT_ASYNC(0);
    }
    __syncthreads();

    V16 a0, a1;
    if (AF32) {
      const float* ap = (const float*)Aall + (size_t)z * sA + arow + k0 + half * 16;
      v8f f0 = *(const v8f*)(ap);
      v8f f1 = *(const v8f*)(ap + 8);
      v8f f2 = *(const v8f*)(ap + 32);
      v8f f3 = *(const v8f*)(ap + 40);
#pragma unroll
      for (int j = 0; j < 8; ++j) {
        a0.h[0][j] = (bf16)f0[j]; a0.h[1][j] = (bf16)f1[j];
        a1.h[0][j] = (bf16)f2[j]; a1.h[1][j] = (bf16)f3[j];
      }
    } else {
      const bf16* ap = (const bf16*)Aall + (size_t)z * sA + arow + k0 + half * 16;
      a0.h[0] = *(const v8bf*)(ap);
      a0.h[1] = *(const v8bf*)(ap + 8);
      a1.h[0] = *(const v8bf*)(ap + 32);
      a1.h[1] = *(const v8bf*)(ap + 40);
    }
#pragma unroll
    for (int f = 0; f < 4; ++f) {
      v16bf b0 = *(const v16bf*)(&Bs[cur][f * 16 + r][half * 16]);
      v16bf b1 = *(const v16bf*)(&Bs[cur][f * 16 + r][32 + half * 16]);
      acc[f] = wmma_bf16(a0.v, b0, acc[f]);
      acc[f] = wmma_bf16(a1.v, b1, acc[f]);
    }
    __syncthreads();
  }

#pragma unroll
  for (int f = 0; f < 4; ++f) {
    const int n    = n0 + f * 16 + r;
    const float bv = bias ? bias[n] : 0.0f;
#pragma unroll
    for (int v = 0; v < 8; ++v) {
      const int m = m0 + v + half * 8;
      float xv = acc[f][v] * alpha + bv;
      if (relu) xv = fmaxf(xv, 0.0f);
      if (outF) outF[(size_t)z * sO + (size_t)m * N + n] = xv;
      if (outB) {
        size_t idx;
        if (mode == 0)
          idx = (size_t)z * sO + (size_t)m * N + n;
        else if (mode == 1)
          idx = ((size_t)(z * NHEAD + (n >> 6)) * M + m) * DH + (n & 63);
        else
          idx = ((size_t)(z * NHEAD + (n >> 6)) * DH + (n & 63)) * (size_t)M + m;
        outB[idx] = (bf16)xv;
      }
    }
  }
}

// ----------------------------------------------------- mask bit-packing ---
// mpack[b][key][qb] bit j = mask[b][qb*8+j][key]
__global__ void k_maskpack(const unsigned char* __restrict__ mask,
                           unsigned char* __restrict__ mpack) {
  const int total = BB * LK * LQB;
  for (int i = blockIdx.x * blockDim.x + threadIdx.x; i < total;
       i += gridDim.x * blockDim.x) {
    const int qb  = i % LQB;
    const int key = (i / LQB) % LK;
    const int b   = i / (LQB * LK);
    unsigned m = 0;
#pragma unroll
    for (int j = 0; j < 8; ++j)
      m |= (unsigned)(mask[((size_t)b * LQ + qb * 8 + j) * LK + key] != 0) << j;
    mpack[i] = (unsigned char)m;
  }
}

// -------------------------------------------------- flash cross-attention --
// 8 waves/block share async-staged K (64x64) and V (64x64) LDS chunks; each
// wave owns a 16-query tile with online softmax over LK in 64-key chunks.
__global__ __launch_bounds__(256) void k_flash(
    const bf16* __restrict__ Q, const bf16* __restrict__ Km,
    const bf16* __restrict__ Vt, const unsigned char* __restrict__ mpack,
    float* __restrict__ out) {
  __shared__ bf16 Ks[2][64][64];   // [key][dh]
  __shared__ bf16 Vs[2][64][64];   // [dh][key]
  __shared__ bf16 Pld[8][16][64];  // per-wave P staging (C-frag -> A-frag)
  const int lane = threadIdx.x & 31;
  const int w    = threadIdx.x >> 5;
  const int r    = lane & 15;
  const int half = lane >> 4;
  const int q0   = (blockIdx.x * 8 + w) * 16;
  const int hh   = blockIdx.y;
  const int b    = blockIdx.z;
  const size_t bh = (size_t)(b * NHEAD + hh);

  auto issue_kv = [&](int kk, int buf) {
    const int t = threadIdx.x;
    // K chunk: 8KB contiguous (64 keys x 128B)
    const unsigned kbase = (unsigned)(uintptr_t)(&Ks[buf][0][0]);
    const unsigned long long kg =
        (unsigned long long)(uintptr_t)(Km + (bh * LK + kk) * DH);
    async_ld16(kbase + t * 16,        kg + t * 16);
    async_ld16(kbase + 4096 + t * 16, kg + 4096 + t * 16);
    // V chunk: 64 dh-rows x 128B
    const int row = t >> 2;
    const int c   = (t & 3) * 32;
    const unsigned vbase = (unsigned)(uintptr_t)(&Vs[buf][0][0]);
    const unsigned long long vg =
        (unsigned long long)(uintptr_t)(Vt + (bh * DH + row) * LK + kk);
    async_ld16(vbase + row * 128 + c,      vg + c);
    async_ld16(vbase + row * 128 + c + 16, vg + c + 16);
  };

  V16 aq[2];  // Q tile 16x64 as two K=32 A-fragments
  {
    const bf16* qp = Q + (bh * LQ + q0 + r) * DH + half * 16;
    aq[0].h[0] = *(const v8bf*)(qp);
    aq[0].h[1] = *(const v8bf*)(qp + 8);
    aq[1].h[0] = *(const v8bf*)(qp + 32);
    aq[1].h[1] = *(const v8bf*)(qp + 40);
  }

  v8f o[4] = {};
  float mrun[8], lrun[8];
#pragma unroll
  for (int v = 0; v < 8; ++v) { mrun[v] = -1e30f; lrun[v] = 0.0f; }
  const unsigned char* mpk = mpack + (size_t)b * LK * LQB + (q0 >> 3) + half;

  issue_kv(0, 0);
  for (int kk = 0; kk < LK; kk += 64) {
    const int cur = (kk >> 6) & 1;
    if (kk + 64 < LK) {
      issue_kv(kk + 64, cur ^ 1);
      WAIT_ASYNC(4);   // this chunk's 4 async ops done; next 4 in flight
    } else {
      WAIT_ASYNC(0);
    }
    __syncthreads();

    // packed mask: one byte covers 8 queries at one key
    unsigned mb[4];
#pragma unroll
    for (int t = 0; t < 4; ++t) mb[t] = mpk[(size_t)(kk + t * 16 + r) * LQB];

    v8f s[4] = {};
#pragma unroll
    for (int kd = 0; kd < 2; ++kd)
#pragma unroll
      for (int t = 0; t < 4; ++t) {
        v16bf bk = *(const v16bf*)(&Ks[cur][t * 16 + r][kd * 32 + half * 16]);
        s[t] = wmma_bf16(aq[kd].v, bk, s[t]);
      }

    // online softmax: C-frag row = v + 8*half, cols live across a 16-lane half
    float pp[4][8];
#pragma unroll
    for (int v = 0; v < 8; ++v) {
      float x[4];
#pragma unroll
      for (int t = 0; t < 4; ++t) {
        x[t] = s[t][v] * 0.125f;            // 1/sqrt(64)
        if ((mb[t] >> v) & 1) x[t] = -1e30f;
      }
      float lm = redmax16(fmaxf(fmaxf(x[0], x[1]), fmaxf(x[2], x[3])));
      const float mn = fmaxf(mrun[v], lm);
      const float sc = __expf(mrun[v] - mn);
      mrun[v] = mn;
      float ls = 0.0f;
#pragma unroll
      for (int t = 0; t < 4; ++t) {
        const float e = __expf(x[t] - mn);
        pp[t][v] = e;
        ls += e;
      }
      ls = redsum16(ls);
      lrun[v] = lrun[v] * sc + ls;
      o[0][v] *= sc; o[1][v] *= sc; o[2][v] *= sc; o[3][v] *= sc;
    }
    // C-frag -> LDS -> A-frag relayout of P (16 queries x 64 keys)
#pragma unroll
    for (int v = 0; v < 8; ++v) {
      const int row = v + half * 8;
#pragma unroll
      for (int t = 0; t < 4; ++t) Pld[w][row][t * 16 + r] = (bf16)pp[t][v];
    }
    asm volatile("s_wait_dscnt 0" ::: "memory");
    V16 ap0, ap1;
    {
      const bf16* pr = &Pld[w][r][0];
      ap0.h[0] = *(const v8bf*)(pr + half * 16);
      ap0.h[1] = *(const v8bf*)(pr + 8 + half * 16);
      ap1.h[0] = *(const v8bf*)(pr + 32 + half * 16);
      ap1.h[1] = *(const v8bf*)(pr + 40 + half * 16);
    }
#pragma unroll
    for (int f = 0; f < 4; ++f) {
      v16bf bv0 = *(const v16bf*)(&Vs[cur][f * 16 + r][half * 16]);
      v16bf bv1 = *(const v16bf*)(&Vs[cur][f * 16 + r][32 + half * 16]);
      o[f] = wmma_bf16(ap0.v, bv0, o[f]);
      o[f] = wmma_bf16(ap1.v, bv1, o[f]);
    }
    __syncthreads();
  }

#pragma unroll
  for (int v = 0; v < 8; ++v) {
    const float inv = lrun[v] > 0.0f ? 1.0f / lrun[v] : 0.0f;
    const size_t base = ((size_t)b * LQ + q0 + v + half * 8) * D_MODEL + hh * DH;
#pragma unroll
    for (int f = 0; f < 4; ++f) out[base + f * 16 + r] = o[f][v] * inv;
  }
}

// ------------------------------------------- residual + LayerNorm (d=512) --
__global__ __launch_bounds__(256) void k_ln(
    const float* __restrict__ base, const float* __restrict__ delta,
    const float* __restrict__ g, const float* __restrict__ be,
    float* __restrict__ outF, bf16* __restrict__ outB) {
  __shared__ float red[2][8];
  const int row = blockIdx.x;
  const int t   = threadIdx.x;
  const size_t off = (size_t)row * D_MODEL;
  float x0 = base[off + t]       + delta[off + t];
  float x1 = base[off + t + 256] + delta[off + t + 256];
  float s  = x0 + x1;
  float ss = x0 * x0 + x1 * x1;
#pragma unroll
  for (int o = 1; o < 32; o <<= 1) {
    s  += __shfl_xor(s, o, 32);
    ss += __shfl_xor(ss, o, 32);
  }
  if ((t & 31) == 0) { red[0][t >> 5] = s; red[1][t >> 5] = ss; }
  __syncthreads();
  float S = 0.0f, SS = 0.0f;
#pragma unroll
  for (int i = 0; i < 8; ++i) { S += red[0][i]; SS += red[1][i]; }
  const float mean = S * (1.0f / D_MODEL);
  const float var  = SS * (1.0f / D_MODEL) - mean * mean;
  const float rstd = rsqrtf(var + 1e-5f);
  const float y0 = (x0 - mean) * rstd * g[t]       + be[t];
  const float y1 = (x1 - mean) * rstd * g[t + 256] + be[t + 256];
  outF[off + t]       = y0;  outB[off + t]       = (bf16)y0;
  outF[off + t + 256] = y1;  outB[off + t + 256] = (bf16)y1;
}

// ------------------------------------------------------------ converts ----
__global__ void k_cvt_bf16(const float* __restrict__ in, bf16* __restrict__ out, int n) {
  for (int i = blockIdx.x * blockDim.x + threadIdx.x; i < n; i += gridDim.x * blockDim.x)
    out[i] = (bf16)in[i];
}
__global__ void k_cvt_both(const float* __restrict__ in, float* __restrict__ outF,
                           bf16* __restrict__ outB, int n) {
  for (int i = blockIdx.x * blockDim.x + threadIdx.x; i < n; i += gridDim.x * blockDim.x) {
    const float v = in[i];
    outF[i] = v;
    outB[i] = (bf16)v;
  }
}
// Wt[n*K+k] = (bf16) W[k*N+n]
__global__ void k_transpose_bf16(const float* __restrict__ W, bf16* __restrict__ Wt,
                                 int K, int N) {
  const int total = K * N;
  for (int i = blockIdx.x * blockDim.x + threadIdx.x; i < total;
       i += gridDim.x * blockDim.x) {
    const int n = i / K;
    const int k = i - n * K;
    Wt[i] = (bf16)W[(size_t)k * N + n];
  }
}

// -------------------------------------------------------------- launch ----
extern "C" void kernel_launch(void* const* d_in, const int* in_sizes, int n_in,
                              void* d_out, int out_size, void* d_ws, size_t ws_size,
                              hipStream_t stream) {
  (void)in_sizes; (void)n_in; (void)out_size; (void)ws_size;
  const float* x    = (const float*)d_in[0];
  const float* h    = (const float*)d_in[1];
  const unsigned char* mask = (const unsigned char*)d_in[2];
  const float* wq   = (const float*)d_in[3];
  const float* wk   = (const float*)d_in[4];
  const float* wv   = (const float*)d_in[5];
  const float* wo   = (const float*)d_in[6];
  const float* ffw1 = (const float*)d_in[7];
  const float* ffb1 = (const float*)d_in[8];
  const float* ffw2 = (const float*)d_in[9];
  const float* ffb2 = (const float*)d_in[10];
  const float* g1   = (const float*)d_in[11];
  const float* be1  = (const float*)d_in[12];
  const float* g2   = (const float*)d_in[13];
  const float* be2  = (const float*)d_in[14];
  const float* owq  = (const float*)d_in[15];
  const float* owk  = (const float*)d_in[16];

  const size_t NXE = (size_t)BB * LK * D_MODEL;  // x-side elements
  const size_t NHE = (size_t)BB * LQ * D_MODEL;  // h-side elements
  const size_t SQ  = (size_t)D_MODEL * D_MODEL;

  char* p = (char*)d_ws;
  auto alloc = [&](size_t bytes) {
    char* q = p;
    p += (bytes + 255) & ~(size_t)255;
    return q;
  };
  bf16*  xbf   = (bf16*)alloc(NXE * 2);
  bf16*  hbf   = (bf16*)alloc(NHE * 2);
  float* hf    = (float*)alloc(NHE * 4);
  bf16*  Qbf   = (bf16*)alloc(NHE * 2);
  bf16*  Kbf   = (bf16*)alloc(NXE * 2);
  bf16*  Vtb   = (bf16*)alloc(NXE * 2);
  float* attnO = (float*)alloc(NHE * 4);
  float* tmp   = (float*)alloc(NHE * 4);
  bf16*  ffbf  = (bf16*)alloc(NHE * 2);
  bf16*  qfbf  = (bf16*)alloc(NHE * 2);
  bf16*  kfbf  = (bf16*)alloc(NXE * 2);
  bf16*  WT    = (bf16*)alloc(20 * SQ * 2);                 // transposed bf16 weights
  unsigned char* mpk = (unsigned char*)alloc((size_t)BB * LK * LQB);

  k_cvt_bf16<<<512, 256, 0, stream>>>(x, xbf, (int)NXE);
  k_cvt_both<<<512, 256, 0, stream>>>(h, hf, hbf, (int)NHE);
  k_maskpack<<<512, 256, 0, stream>>>(mask, mpk);
  for (int l = 0; l < 3; ++l) {
    k_transpose_bf16<<<256, 256, 0, stream>>>(wq   + l * SQ, WT + (size_t)(l * 6 + 0) * SQ, 512, 512);
    k_transpose_bf16<<<256, 256, 0, stream>>>(wk   + l * SQ, WT + (size_t)(l * 6 + 1) * SQ, 512, 512);
    k_transpose_bf16<<<256, 256, 0, stream>>>(wv   + l * SQ, WT + (size_t)(l * 6 + 2) * SQ, 512, 512);
    k_transpose_bf16<<<256, 256, 0, stream>>>(wo   + l * SQ, WT + (size_t)(l * 6 + 3) * SQ, 512, 512);
    k_transpose_bf16<<<256, 256, 0, stream>>>(ffw1 + l * SQ, WT + (size_t)(l * 6 + 4) * SQ, 512, 512);
    k_transpose_bf16<<<256, 256, 0, stream>>>(ffw2 + l * SQ, WT + (size_t)(l * 6 + 5) * SQ, 512, 512);
  }
  k_transpose_bf16<<<256, 256, 0, stream>>>(owq, WT + (size_t)18 * SQ, 512, 512);
  k_transpose_bf16<<<256, 256, 0, stream>>>(owk, WT + (size_t)19 * SQ, 512, 512);

  auto gemm = [&](const void* A, bool af32, const bf16* Bt, const float* bias,
                  float* oF, bf16* oB, int M, int N, int K, long sA, long sBt,
                  long sO, int mode, int relu, float alpha) {
    dim3 gr(N / 64, M / 128, BB);
    if (af32)
      k_gemm<true><<<gr, 256, 0, stream>>>(A, Bt, bias, oF, oB, M, N, K, sA, sBt, sO, mode, relu, alpha);
    else
      k_gemm<false><<<gr, 256, 0, stream>>>(A, Bt, bias, oF, oB, M, N, K, sA, sBt, sO, mode, relu, alpha);
  };

  for (int l = 0; l < 3; ++l) {
    const bf16* WqT = WT + (size_t)(l * 6 + 0) * SQ;
    const bf16* WkT = WT + (size_t)(l * 6 + 1) * SQ;
    const bf16* WvT = WT + (size_t)(l * 6 + 2) * SQ;
    const bf16* WoT = WT + (size_t)(l * 6 + 3) * SQ;
    const bf16* W1T = WT + (size_t)(l * 6 + 4) * SQ;
    const bf16* W2T = WT + (size_t)(l * 6 + 5) * SQ;

    gemm(hbf, false, WqT, nullptr, nullptr, Qbf, LQ, 512, 512, (long)LQ * 512, 0, 0, 1, 0, 1.0f);
    gemm(xbf, false, WkT, nullptr, nullptr, Kbf, LK, 512, 512, (long)LK * 512, 0, 0, 1, 0, 1.0f);
    gemm(xbf, false, WvT, nullptr, nullptr, Vtb, LK, 512, 512, (long)LK * 512, 0, 0, 2, 0, 1.0f);
    k_flash<<<dim3(LQ / 128, NHEAD, BB), 256, 0, stream>>>(Qbf, Kbf, Vtb, mpk, attnO);
    gemm(attnO, true, WoT, nullptr, tmp, nullptr, LQ, 512, 512, (long)LQ * 512, 0, (long)LQ * 512, 0, 0, 1.0f);
    k_ln<<<BB * LQ, 256, 0, stream>>>(hf, tmp, g1 + l * 512, be1 + l * 512, hf, hbf);
    gemm(hbf, false, W1T, ffb1 + l * 512, nullptr, ffbf, LQ, 512, 512, (long)LQ * 512, 0, (long)LQ * 512, 0, 1, 1.0f);
    gemm(ffbf, false, W2T, ffb2 + l * 512, tmp, nullptr, LQ, 512, 512, (long)LQ * 512, 0, (long)LQ * 512, 0, 0, 1.0f);
    k_ln<<<BB * LQ, 256, 0, stream>>>(hf, tmp, g2 + l * 512, be2 + l * 512, hf, hbf);
  }

  // final: u = (h @ out_wq) @ (x @ out_wk)^T / sqrt(512)
  gemm(hbf, false, WT + (size_t)18 * SQ, nullptr, nullptr, qfbf, LQ, 512, 512,
       (long)LQ * 512, 0, (long)LQ * 512, 0, 0, 1.0f);
  gemm(xbf, false, WT + (size_t)19 * SQ, nullptr, nullptr, kfbf, LK, 512, 512,
       (long)LK * 512, 0, (long)LK * 512, 0, 0, 1.0f);
  gemm(qfbf, false, kfbf, nullptr, (float*)d_out, nullptr, LQ, LK, 512,
       (long)LQ * 512, (long)LK * 512, (long)LQ * LK, 0, 0, 0.044194173824159216f);
}
